// MSCLoss_84971632984673
// MI455X (gfx1250) — matrix operand order repaired and checked by reading, not compiled, not run on hardware
//
#include <hip/hip_runtime.h>
#include <hip/hip_bf16.h>

// ---------------------------------------------------------------------------
// MSC loss pipeline for MI455X (gfx1250, wave32, WMMA).
//
// Stages:
//   P  prep:      row L2 norms (reciprocal) + f32->f16 feature copies
//   G  gemm x2:   simT[j][i]  = <t_j, s_i> * rn_t[j]*rn_s[i]   (WMMA f16->f32)
//                 sim0T[j][i] = <t0_j, s_i>* rn_t0[j]*rn_s[i]
//                 (TARGET-major so all later column ops are contiguous rows)
//   S  per-column bitonic sort (LDS), top-10 vote -> assigned, num_correct,
//                 masked top-20 ratio -> scores[j]
//   K  top-2048 of scores (single-block bitonic)
//   L  per-selected-column softmax contrast -> colLoss[c]
//   F  deterministic final reduction -> d_out[0]=loss, d_out[1]=num_correct
// ---------------------------------------------------------------------------

#define NS 4096          // sources
#define NT 4096          // targets
#define DIM 512
#define RANKING_K 20
#define TOP_RANKED_N 2048
#define TOP_N_SIM 10
#define TAU 0.05f
#define NCLS 64
#define EPS_LOG 1e-6f

typedef __attribute__((ext_vector_type(16))) _Float16 v16h;
typedef __attribute__((ext_vector_type(8)))  _Float16 v8h;
typedef __attribute__((ext_vector_type(8)))  float    v8f;

// ---------------------------------------------------------------- stage P ---
__global__ __launch_bounds__(128) void prep_kernel(
    const float* __restrict__ src, const float* __restrict__ tgt,
    const float* __restrict__ tgt0,
    _Float16* __restrict__ sH, _Float16* __restrict__ tH, _Float16* __restrict__ t0H,
    float* __restrict__ rnS, float* __restrict__ rnT, float* __restrict__ rnT0)
{
    const int row   = blockIdx.x & (NS - 1);
    const int which = blockIdx.x >> 12;          // 0=src 1=tgt 2=tgt0
    const float* in = (which == 0) ? src : (which == 1) ? tgt : tgt0;
    _Float16* out   = (which == 0) ? sH  : (which == 1) ? tH  : t0H;
    float* rn       = (which == 0) ? rnS : (which == 1) ? rnT : rnT0;

    float ss = 0.0f;
    for (int c = threadIdx.x; c < DIM; c += 128) {
        float x = in[row * DIM + c];
        out[row * DIM + c] = (_Float16)x;
        ss += x * x;
    }
    __shared__ float red[128];
    red[threadIdx.x] = ss;
    __syncthreads();
    for (int st = 64; st > 0; st >>= 1) {
        if (threadIdx.x < st) red[threadIdx.x] += red[threadIdx.x + st];
        __syncthreads();
    }
    if (threadIdx.x == 0) rn[row] = rsqrtf(red[0]);
}

// ---------------------------------------------------------------- stage G ---
// C[m][n] = (A_m . B_n) * rnA[m] * rnB[n],  A,B row-major [4096][512] f16.
// One 16x16 tile per wave, K-loop of 16 x v_wmma_f32_16x16x32_f16.
__global__ __launch_bounds__(256) void gemm_cos_wmma(
    const _Float16* __restrict__ A, const _Float16* __restrict__ B,
    const float* __restrict__ rnA, const float* __restrict__ rnB,
    float* __restrict__ C)
{
    const int lane  = threadIdx.x & 31;
    const int wave  = threadIdx.x >> 5;
    const int tile  = blockIdx.x * 8 + wave;     // 65536 tiles total
    const int ti    = tile & 255;                // M tile
    const int tj    = tile >> 8;                 // N tile
    const int ibase = ti * 16, jbase = tj * 16;
    const int lmod  = lane & 15, lhalf = lane >> 4;

    const _Float16* arow = A + (size_t)(ibase + lmod) * DIM;
    const _Float16* brow = B + (size_t)(jbase + lmod) * DIM;

    v8f acc = {};
#pragma unroll
    for (int kb = 0; kb < DIM; kb += 32) {
        // A fragment (ISA 16-bit A 16x32 layout): lane m=L%16,
        // halves 0..7 -> K = kb + (L/16)*8 + h, halves 8..15 -> +16
        v8h a0 = *(const v8h*)(arow + kb + lhalf * 8);
        v8h a1 = *(const v8h*)(arow + kb + 16 + lhalf * 8);
        // B fragment: lane n=L%16, same K mapping as A
        v8h b0 = *(const v8h*)(brow + kb + lhalf * 8);
        v8h b1 = *(const v8h*)(brow + kb + 16 + lhalf * 8);
        v16h a, b;
#pragma unroll
        for (int h = 0; h < 8; ++h) {
            a[h] = a0[h]; a[h + 8] = a1[h];
            b[h] = b0[h]; b[h + 8] = b1[h];
        }
        acc = __builtin_amdgcn_wmma_f32_16x16x32_f16(
            /*neg_a=*/false, a, /*neg_b=*/false, b,
            /*c_mod=*/(short)0, acc, /*reuse_a=*/false, /*reuse_b=*/false);
    }

    // D layout: VGPR r, lane L -> M = r + (L/16)*8, N = L%16
    const int jj = jbase + lmod;
    const float rb = rnB[jj];
#pragma unroll
    for (int r = 0; r < 8; ++r) {
        const int i = ibase + r + lhalf * 8;
        C[(size_t)i * NT + jj] = acc[r] * rnA[i] * rb;
    }
}

// -------------------------------------------------- bitonic sort (LDS) -----
// Descending by value, ascending index on ties. n power of two.
__device__ __forceinline__ void bitonic_sort_desc(float* v, int* ix, int n,
                                                  int tid, int nthr)
{
    for (int k = 2; k <= n; k <<= 1) {
        for (int j = k >> 1; j > 0; j >>= 1) {
            __syncthreads();
            for (int i = tid; i < n; i += nthr) {
                int l = i ^ j;
                if (l > i) {
                    float vi = v[i], vl = v[l];
                    int xi = ix[i], xl = ix[l];
                    bool dirDesc = ((i & k) == 0);
                    bool sw = dirDesc ? (vi < vl || (vi == vl && xi > xl))
                                      : (vi > vl || (vi == vl && xi < xl));
                    if (sw) { v[i] = vl; v[l] = vi; ix[i] = xl; ix[l] = xi; }
                }
            }
        }
    }
    __syncthreads();
}

// ---------------------------------------------------------------- stage S ---
// One block per target column j: sort simT row j, vote, ratio score.
__global__ __launch_bounds__(256) void column_kernel(
    const float* __restrict__ simT, const float* __restrict__ sim0T,
    const int* __restrict__ srcLab, const int* __restrict__ tgtLab,
    int* __restrict__ assigned, float* __restrict__ scores,
    int* __restrict__ numCorrect)
{
    __shared__ float sv[NS];
    __shared__ int   si[NS];
    __shared__ int   cnt[NCLS];
    __shared__ int   pm[256], pn[256];
    __shared__ float rM[256], rN[256];
    __shared__ int   assignedS;

    const int j   = blockIdx.x;
    const int tid = threadIdx.x;
    const float* simRow  = simT  + (size_t)j * NS;
    const float* sim0Row = sim0T + (size_t)j * NS;

    for (int i = tid; i < NS; i += 256) { sv[i] = simRow[i]; si[i] = i; }
    bitonic_sort_desc(sv, si, NS, tid, 256);

    // top-10 majority vote
    if (tid < NCLS) cnt[tid] = 0;
    __syncthreads();
    if (tid < TOP_N_SIM) atomicAdd(&cnt[srcLab[si[tid]]], 1);
    __syncthreads();
    if (tid == 0) {
        int best = -1, bc = 0;
        for (int c = 0; c < NCLS; ++c)       // first max, like jnp.argmax
            if (cnt[c] > best) { best = cnt[c]; bc = c; }
        assignedS = bc;
        assigned[j] = bc;
        if (bc == tgtLab[j]) atomicAdd(numCorrect, 1);
    }
    __syncthreads();
    const int am = assignedS;

    // masked top-20 sums over sim0 in sorted order (block-wide prefix counts)
    const int CH = NS / 256;
    const int base = tid * CH;
    int cm = 0, cn = 0;
    for (int r = 0; r < CH; ++r) {
        if (srcLab[si[base + r]] == am) ++cm; else ++cn;
    }
    pm[tid] = cm; pn[tid] = cn;
    __syncthreads();
    if (tid == 0) {
        int sm = 0, sn = 0;
        for (int t = 0; t < 256; ++t) {
            int a = pm[t]; pm[t] = sm; sm += a;
            int b = pn[t]; pn[t] = sn; sn += b;
        }
    }
    __syncthreads();
    cm = pm[tid]; cn = pn[tid];
    float sM = 0.0f, sN = 0.0f;
    for (int r = 0; r < CH; ++r) {
        const int sidx = si[base + r];
        if (srcLab[sidx] == am) {
            if (++cm <= RANKING_K) sM += sim0Row[sidx];
        } else {
            if (++cn <= RANKING_K) sN += sim0Row[sidx];
        }
    }
    rM[tid] = sM; rN[tid] = sN;
    __syncthreads();
    for (int st = 128; st > 0; st >>= 1) {
        if (tid < st) { rM[tid] += rM[tid + st]; rN[tid] += rN[tid + st]; }
        __syncthreads();
    }
    if (tid == 0) scores[j] = rM[0] / rN[0];
}

// ---------------------------------------------------------------- stage K ---
__global__ __launch_bounds__(256) void topk_kernel(
    const float* __restrict__ scores, int* __restrict__ topIdx)
{
    __shared__ float sv[NT];
    __shared__ int   si[NT];
    const int tid = threadIdx.x;
    for (int i = tid; i < NT; i += 256) { sv[i] = scores[i]; si[i] = i; }
    bitonic_sort_desc(sv, si, NT, tid, 256);
    for (int i = tid; i < TOP_RANKED_N; i += 256) topIdx[i] = si[i];
}

// ---------------------------------------------------------------- stage L ---
__global__ __launch_bounds__(256) void contrast_kernel(
    const float* __restrict__ sim0T, const int* __restrict__ topIdx,
    const int* __restrict__ assigned, const int* __restrict__ srcLab,
    float* __restrict__ colLoss)
{
    __shared__ float red0[256], red1[256];
    const int c   = blockIdx.x;
    const int tid = threadIdx.x;
    const int j   = topIdx[c];
    const int cl  = assigned[j];
    const float* row = sim0T + (size_t)j * NS;

    float mx = -3.4e38f;
    for (int i = tid; i < NS; i += 256) mx = fmaxf(mx, row[i]);
    red0[tid] = mx;
    __syncthreads();
    for (int st = 128; st > 0; st >>= 1) {
        if (tid < st) red0[tid] = fmaxf(red0[tid], red0[tid + st]);
        __syncthreads();
    }
    const float M = red0[0];
    __syncthreads();

    float num = 0.0f, den = 0.0f;
    for (int i = tid; i < NS; i += 256) {
        float e = __expf((row[i] - M) * (1.0f / TAU));
        den += e;
        if (srcLab[i] == cl) num += e;
    }
    red0[tid] = num; red1[tid] = den;
    __syncthreads();
    for (int st = 128; st > 0; st >>= 1) {
        if (tid < st) { red0[tid] += red0[tid + st]; red1[tid] += red1[tid + st]; }
        __syncthreads();
    }
    if (tid == 0) colLoss[c] = -logf(red0[0] / red1[0] + EPS_LOG);
}

// ------------------------------------------------------------ init / final --
__global__ void init_kernel(int* __restrict__ numCorrect)
{
    if (threadIdx.x == 0) *numCorrect = 0;
}

__global__ __launch_bounds__(256) void finalize_kernel(
    const float* __restrict__ colLoss, const int* __restrict__ numCorrect,
    float* __restrict__ out)
{
    __shared__ float red[256];
    const int tid = threadIdx.x;
    float s = 0.0f;
    for (int i = tid; i < TOP_RANKED_N; i += 256) s += colLoss[i];
    red[tid] = s;
    __syncthreads();
    for (int st = 128; st > 0; st >>= 1) {
        if (tid < st) red[tid] += red[tid + st];
        __syncthreads();
    }
    if (tid == 0) {
        out[0] = red[0] / (float)TOP_RANKED_N;
        out[1] = (float)(*numCorrect);
    }
}

// ---------------------------------------------------------------------------
extern "C" void kernel_launch(void* const* d_in, const int* in_sizes, int n_in,
                              void* d_out, int out_size, void* d_ws, size_t ws_size,
                              hipStream_t stream)
{
    const float* srcF  = (const float*)d_in[0];
    const int*   srcL  = (const int*)  d_in[1];
    const float* tgtF  = (const float*)d_in[2];
    const float* tgt0F = (const float*)d_in[3];
    const int*   tgtL  = (const int*)  d_in[4];
    float* out = (float*)d_out;

    // workspace carve-out (~147 MB total)
    char* ws = (char*)d_ws;
    size_t off = 0;
    auto carve = [&](size_t bytes) {
        char* p = ws + off;
        off = (off + bytes + 255) & ~(size_t)255;
        return p;
    };
    float*    simT   = (float*)   carve((size_t)NT * NS * 4);   // [tgt][src]
    float*    sim0T  = (float*)   carve((size_t)NT * NS * 4);   // [tgt][src]
    _Float16* sH     = (_Float16*)carve((size_t)NS * DIM * 2);
    _Float16* tH     = (_Float16*)carve((size_t)NT * DIM * 2);
    _Float16* t0H    = (_Float16*)carve((size_t)NT * DIM * 2);
    float*    rnS    = (float*)   carve(NS * 4);
    float*    rnT    = (float*)   carve(NT * 4);
    float*    rnT0   = (float*)   carve(NT * 4);
    int*      assigned = (int*)   carve(NT * 4);
    float*    scores   = (float*) carve(NT * 4);
    int*      topIdx   = (int*)   carve(TOP_RANKED_N * 4);
    float*    colLoss  = (float*) carve(TOP_RANKED_N * 4);
    int*      nCorrect = (int*)   carve(4);
    (void)ws_size; (void)in_sizes; (void)n_in; (void)out_size;

    init_kernel<<<1, 32, 0, stream>>>(nCorrect);
    prep_kernel<<<3 * NS, 128, 0, stream>>>(srcF, tgtF, tgt0F,
                                            sH, tH, t0H, rnS, rnT, rnT0);
    // target-major: A = target rows (M), B = source rows (N)
    gemm_cos_wmma<<<(NT / 16) * (NS / 16) / 8, 256, 0, stream>>>(
        tH, sH, rnT, rnS, simT);
    gemm_cos_wmma<<<(NT / 16) * (NS / 16) / 8, 256, 0, stream>>>(
        t0H, sH, rnT0, rnS, sim0T);
    column_kernel<<<NT, 256, 0, stream>>>(simT, sim0T, srcL, tgtL,
                                          assigned, scores, nCorrect);
    topk_kernel<<<1, 256, 0, stream>>>(scores, topIdx);
    contrast_kernel<<<TOP_RANKED_N, 256, 0, stream>>>(sim0T, topIdx, assigned,
                                                      srcL, colLoss);
    finalize_kernel<<<1, 256, 0, stream>>>(colLoss, nCorrect, out);
}
